// ScatterAggregator_60172491817653
// MI455X (gfx1250) — compile-verified, matrix operand order checked by reading, not run on hardware
//
#include <hip/hip_runtime.h>

#define NPIX      12288
#define NBUCKETS  32
#define IN_DIM    32
#define PROJ_IN   1056   // 32*32 + 32
#define PROJ_OUT  256
#define OUT_DIM   128
#define LN_EPS    1e-5f

typedef __attribute__((ext_vector_type(16))) __bf16          v16bf;
typedef __attribute__((ext_vector_type(8)))  float           v8f;
typedef __attribute__((ext_vector_type(4)))  float           f4;
typedef __attribute__((ext_vector_type(4)))  unsigned short  us4;

// LDS layout of the fused kernel (dynamic shared, 100352 B):
//   [0      .. 67584)  ldsA : 2 M-tiles x (33 kb x 32 lanes x 16 halves) bf16
//   [67584  ..100352)  hbuf : 32 x 256 f32
//   ldsH (2 x 8x32x16 bf16 = 16 KB) overlays ldsA once GEMM1 is done.
#define A_TILE_HALVES (33 * 32 * 16)
#define H_TILE_HALVES (8 * 32 * 16)
#define SMEM_A_BYTES  (2 * A_TILE_HALVES * 2)          // 67584
#define SMEM_BYTES    (SMEM_A_BYTES + 32 * PROJ_OUT * 4) // 100352

__device__ __forceinline__ unsigned short f32_to_bf16_bits(float f) {
    union { float f; unsigned int u; } c; c.f = f;
    unsigned int u = c.u;
    u += 0x7FFFu + ((u >> 16) & 1u);   // round-to-nearest-even
    return (unsigned short)(u >> 16);
}

// Fragment-swizzled position for a 16-bit WMMA A-matrix element (row m, k).
// ISA 16-bit A 16x32 layout: lanes 0-15 hold K=0..7 (V0..V3) and K=16..23
// (V4..V7); lanes 16-31 hold K=8..15 and K=24..31. Each lane's 16 halves are
// stored contiguously so fragments load as two b128s. Any 4-aligned run of 4
// consecutive K values lands on 4 consecutive halves (one b64 store).
__device__ __forceinline__ int a_swz_pos(int m, int k) {
    int kb = k >> 5, kk = k & 31;
    int group  = kk >> 4;
    int rem    = kk & 15;
    int half   = rem >> 3;
    int rem2   = rem & 7;
    int within = rem2 >> 1;
    int lo     = rem2 & 1;
    int lane   = half * 16 + m;
    int e      = (group * 4 + within) * 2 + lo;
    return (kb * 32 + lane) * 16 + e;
}

// ---------------------------------------------------------------- zero fill
__global__ void zero_f4_kernel(float4* __restrict__ p, long long n4) {
    long long i = (long long)blockIdx.x * blockDim.x + threadIdx.x;
    long long stride = (long long)gridDim.x * blockDim.x;
    float4 z = make_float4(0.f, 0.f, 0.f, 0.f);
    for (; i < n4; i += stride) p[i] = z;
}

// ------------------------------------------- W1/W2 -> bf16 B-fragment order
__global__ void swz_w1_kernel(const float* __restrict__ W1, unsigned short* __restrict__ dst) {
    int idx = blockIdx.x * blockDim.x + threadIdx.x;
    if (idx >= 33 * 16 * 32 * 16) return;            // 33 K-chunks, 16 N-tiles
    int e = idx & 15, lane = (idx >> 4) & 31, ntile = (idx >> 9) & 15, kb = idx >> 13;
    int half = lane >> 4, ncol = ntile * 16 + (lane & 15);
    int j = e >> 1, lo = e & 1, group = j >> 2, within = j & 3;
    int k = kb * 32 + group * 16 + half * 8 + within * 2 + lo;
    dst[idx] = f32_to_bf16_bits(W1[k * PROJ_OUT + ncol]);
}

__global__ void swz_w2_kernel(const float* __restrict__ W2, unsigned short* __restrict__ dst) {
    int idx = blockIdx.x * blockDim.x + threadIdx.x;
    if (idx >= 8 * 8 * 32 * 16) return;              // 8 K-chunks, 8 N-tiles
    int e = idx & 15, lane = (idx >> 4) & 31, ntile = (idx >> 9) & 7, kb = idx >> 12;
    int half = lane >> 4, ncol = ntile * 16 + (lane & 15);
    int j = e >> 1, lo = e & 1, group = j >> 2, within = j & 3;
    int k = kb * 32 + group * 16 + half * 8 + within * 2 + lo;
    dst[idx] = f32_to_bf16_bits(W2[k * OUT_DIM + ncol]);
}

// -------------------------------------------------------------- scatter-add
// One wave per observation row: lane c atomically adds channel c (one 128 B
// coalesced atomic burst); lane 0 bumps the count.
__global__ void scatter_kernel(const float* __restrict__ obs,
                               const int*   __restrict__ bidx,
                               const int*   __restrict__ pix,
                               const int*   __restrict__ bucket,
                               float* __restrict__ sums,
                               float* __restrict__ counts,
                               int nrows) {
    long long t = (long long)blockIdx.x * blockDim.x + threadIdx.x;
    int row = (int)(t >> 5);
    int c   = (int)(t & 31);
    if (row >= nrows) return;
    int flat = bidx[row] * (NPIX * NBUCKETS) + pix[row] * NBUCKETS + bucket[row];
    float v = __builtin_nontemporal_load(&obs[(long long)row * IN_DIM + c]);
    atomicAdd(&sums[(long long)flat * IN_DIM + c], v);
    if (c == 0) atomicAdd(&counts[flat], 1.0f);
}

// -------------------------------------------------- fused mean+MLP+LN+SiLU
// One workgroup (8 waves, 256 thr) handles 32 pixel-rows (2 M-tiles) so each
// W1 B-fragment load feeds 2 WMMAs (halves L2 weight traffic vs 16 rows).
__global__ __launch_bounds__(256)
void fused_mlp_kernel(const float* __restrict__ sums,
                      const float* __restrict__ counts,
                      const unsigned short* __restrict__ w1swz,
                      const unsigned short* __restrict__ w2swz,
                      const float* __restrict__ b1,
                      const float* __restrict__ lng,
                      const float* __restrict__ lnb,
                      const float* __restrict__ b2,
                      float* __restrict__ out) {
    extern __shared__ char smem[];
    unsigned short* ldsA = (unsigned short*)smem;                 // 2 A tiles
    float*          hbuf = (float*)(smem + SMEM_A_BYTES);         // 32 x 256
    unsigned short* ldsH = (unsigned short*)smem;                 // overlays ldsA

    const int t    = threadIdx.x;
    const int lane = t & 31;
    const int wave = t >> 5;
    const long long gr0 = (long long)blockIdx.x * 32;

    // ---- Phase 1: scatter-mean + has_obs -> bf16 A tiles (swizzled) -------
    {
        int m   = t >> 3;               // row 0..31
        int seg = t & 7;                // 8 threads per row
        int mt  = m >> 4, mrow = m & 15;
        long long gr = gr0 + m;
        const f4*    srow4 = (const f4*)(sums + gr * 1024);
        const float* crow  = counts + gr * 32;
        unsigned short* Adst = ldsA + mt * A_TILE_HALVES;
        #pragma unroll 4
        for (int i = 0; i < 32; ++i) {
            int col4 = seg + 8 * i;                 // float4 index 0..255
            f4 v = __builtin_nontemporal_load(&srow4[col4]);
            float cnt = crow[col4 >> 3];            // one bucket per quad
            float inv = (cnt > 0.0f) ? 1.0f / cnt : 0.0f;
            us4 pk;
            pk[0] = f32_to_bf16_bits(v[0] * inv);
            pk[1] = f32_to_bf16_bits(v[1] * inv);
            pk[2] = f32_to_bf16_bits(v[2] * inv);
            pk[3] = f32_to_bf16_bits(v[3] * inv);
            *(us4*)&Adst[a_swz_pos(mrow, col4 * 4)] = pk;
        }
        {   // has_obs flags: cols 1024..1055, 4 per thread, one b64 store
            int c0 = seg * 4;
            us4 pk;
            #pragma unroll
            for (int jj = 0; jj < 4; ++jj)
                pk[jj] = (crow[c0 + jj] > 0.0f) ? (unsigned short)0x3F80 : (unsigned short)0;
            *(us4*)&Adst[a_swz_pos(mrow, 1024 + c0)] = pk;
        }
    }
    __syncthreads();

    // ---- Phase 2: GEMM1 (32x1056 @ 1056x256): wave owns 2 N-tiles x 2 M ---
    {
        v8f a00 = {}, a01 = {}, a10 = {}, a11 = {};   // [mt][nt]
        const int nt0 = wave * 2, nt1 = wave * 2 + 1;
        const v16bf* ApA = (const v16bf*)ldsA;
        const v16bf* ApB = ApA + 33 * 32;             // second M-tile
        const v16bf* Bp  = (const v16bf*)w1swz;
        for (int kb = 0; kb < 33; ++kb) {
            v16bf fa0 = ApA[kb * 32 + lane];
            v16bf fa1 = ApB[kb * 32 + lane];
            v16bf fb0 = Bp[(kb * 16 + nt0) * 32 + lane];
            v16bf fb1 = Bp[(kb * 16 + nt1) * 32 + lane];
            a00 = __builtin_amdgcn_wmma_f32_16x16x32_bf16(false, fa0, false, fb0, (short)0, a00, false, false);
            a01 = __builtin_amdgcn_wmma_f32_16x16x32_bf16(false, fa0, false, fb1, (short)0, a01, false, false);
            a10 = __builtin_amdgcn_wmma_f32_16x16x32_bf16(false, fa1, false, fb0, (short)0, a10, false, false);
            a11 = __builtin_amdgcn_wmma_f32_16x16x32_bf16(false, fa1, false, fb1, (short)0, a11, false, false);
        }
        int mhi = (lane >> 4) * 8;
        int n0 = nt0 * 16 + (lane & 15);
        int n1 = nt1 * 16 + (lane & 15);
        float bias0 = b1[n0], bias1 = b1[n1];
        #pragma unroll
        for (int i = 0; i < 8; ++i) {
            hbuf[(mhi + i) * PROJ_OUT + n0]      = a00[i] + bias0;
            hbuf[(mhi + i) * PROJ_OUT + n1]      = a01[i] + bias1;
            hbuf[(16 + mhi + i) * PROJ_OUT + n0] = a10[i] + bias0;
            hbuf[(16 + mhi + i) * PROJ_OUT + n1] = a11[i] + bias1;
        }
    }
    __syncthreads();

    // ---- Phase 3: LayerNorm + SiLU -> bf16 (swizzled), overlaying ldsA ----
    {
        int m = t >> 3, seg = t & 7;                  // 8 threads per row
        int mt = m >> 4, mrow = m & 15;
        const f4* hrow4 = (const f4*)(hbuf + m * PROJ_OUT);
        float x[32];
        float s1 = 0.f, s2 = 0.f;
        #pragma unroll
        for (int j = 0; j < 8; ++j) {
            f4 v = hrow4[seg * 8 + j];
            #pragma unroll
            for (int e2 = 0; e2 < 4; ++e2) {
                float xv = v[e2];
                x[j * 4 + e2] = xv;
                s1 += xv;
                s2 += xv * xv;
            }
        }
        for (int off = 1; off < 8; off <<= 1) {       // 8-wide reduction
            s1 += __shfl_xor(s1, off, 8);
            s2 += __shfl_xor(s2, off, 8);
        }
        float mu  = s1 * (1.0f / 256.0f);
        float var = s2 * (1.0f / 256.0f) - mu * mu;
        float rs  = rsqrtf(var + LN_EPS);
        unsigned short* Hdst = ldsH + mt * H_TILE_HALVES;
        #pragma unroll
        for (int j = 0; j < 8; ++j) {
            int k0 = seg * 32 + j * 4;
            us4 pk;
            #pragma unroll
            for (int e2 = 0; e2 < 4; ++e2) {
                int col = k0 + e2;
                float y   = (x[j * 4 + e2] - mu) * rs * lng[col] + lnb[col];
                float sil = y / (1.0f + __expf(-y));
                pk[e2] = f32_to_bf16_bits(sil);
            }
            *(us4*)&Hdst[a_swz_pos(mrow, k0)] = pk;
        }
    }
    __syncthreads();

    // ---- Phase 4: GEMM2 (32x256 @ 256x128): wave owns 1 N-tile x 2 M ------
    {
        v8f acc0 = {}, acc1 = {};
        const v16bf* ApA = (const v16bf*)ldsH;
        const v16bf* ApB = ApA + 8 * 32;
        const v16bf* Bp  = (const v16bf*)w2swz;
        for (int kb = 0; kb < 8; ++kb) {
            v16bf fa0 = ApA[kb * 32 + lane];
            v16bf fa1 = ApB[kb * 32 + lane];
            v16bf fb  = Bp[(kb * 8 + wave) * 32 + lane];
            acc0 = __builtin_amdgcn_wmma_f32_16x16x32_bf16(false, fa0, false, fb, (short)0, acc0, false, false);
            acc1 = __builtin_amdgcn_wmma_f32_16x16x32_bf16(false, fa1, false, fb, (short)0, acc1, false, false);
        }
        int mhi = (lane >> 4) * 8;
        int n   = wave * 16 + (lane & 15);
        float bias = b2[n];
        #pragma unroll
        for (int i = 0; i < 8; ++i) {
            __builtin_nontemporal_store(acc0[i] + bias, &out[(gr0 + mhi + i) * OUT_DIM + n]);
            __builtin_nontemporal_store(acc1[i] + bias, &out[(gr0 + 16 + mhi + i) * OUT_DIM + n]);
        }
    }
}

extern "C" void kernel_launch(void* const* d_in, const int* in_sizes, int n_in,
                              void* d_out, int out_size, void* d_ws, size_t ws_size,
                              hipStream_t stream) {
    const float* obs    = (const float*)d_in[0];
    const int*   bidx   = (const int*)  d_in[1];
    const int*   pix    = (const int*)  d_in[2];
    const int*   bucket = (const int*)  d_in[3];
    // d_in[4] = nbatch scalar on device (unreadable during capture); derive B from out_size
    const float* W1  = (const float*)d_in[5];
    const float* b1  = (const float*)d_in[6];
    const float* lng = (const float*)d_in[7];
    const float* lnb = (const float*)d_in[8];
    const float* W2  = (const float*)d_in[9];
    const float* b2  = (const float*)d_in[10];
    float* out = (float*)d_out;

    const int nrows = in_sizes[0] / IN_DIM;                 // 2,000,000
    const int B     = out_size / (NPIX * OUT_DIM);          // 4
    const long long rows_total = (long long)B * NPIX;       // 49152
    const long long cells      = rows_total * NBUCKETS;     // 1,572,864

    const size_t sums_f   = (size_t)cells * IN_DIM;         // 50,331,648 floats
    const size_t counts_f = (size_t)cells;
    const size_t w1_half  = (size_t)33 * 16 * 32 * 16;      // 270,336 bf16
    const size_t w2_half  = (size_t)8  * 8  * 32 * 16;      //  32,768 bf16
    const size_t need = (sums_f + counts_f) * 4 + (w1_half + w2_half) * 2;
    if (ws_size < need) return;                             // deterministic no-op guard

    char* ws = (char*)d_ws;
    float* sums   = (float*)ws;
    float* counts = (float*)(ws + sums_f * 4);
    unsigned short* w1swz = (unsigned short*)(ws + (sums_f + counts_f) * 4);
    unsigned short* w2swz = w1swz + w1_half;

    // 1) zero the sums + counts grid (~208 MB)
    long long n4 = (long long)(sums_f + counts_f) / 4;
    zero_f4_kernel<<<4096, 256, 0, stream>>>((float4*)ws, n4);

    // 2) weights -> bf16 fragment order (tiny, L2-resident afterwards)
    swz_w1_kernel<<<(int)((w1_half + 255) / 256), 256, 0, stream>>>(W1, w1swz);
    swz_w2_kernel<<<(int)((w2_half + 255) / 256), 256, 0, stream>>>(W2, w2swz);

    // 3) scatter-add: one lane per (row, channel)
    long long sthreads = (long long)nrows * 32;
    int sblocks = (int)((sthreads + 255) / 256);
    scatter_kernel<<<sblocks, 256, 0, stream>>>(obs, bidx, pix, bucket, sums, counts, nrows);

    // 4) fused mean + GEMM1 + LN + SiLU + GEMM2 (32 rows per workgroup)
    fused_mlp_kernel<<<(int)(rows_total / 32), 256, SMEM_BYTES, stream>>>(
        sums, counts, w1swz, w2swz, b1, lng, lnb, b2, out);
}
